// PointNet_69947837383384
// MI455X (gfx1250) — compile-verified, matrix operand order checked by reading, not run on hardware
//
#include <hip/hip_runtime.h>

// ---------------------------------------------------------------------------
// PointNet fused inference for MI455X (gfx1250, wave32, WMMA bf16).
//
//   * layer5 concat split:  z5 = relu(W5[:, :64] @ h2 + c5),
//     c5 = W5[:, 64:] @ g + b5   (tiny GEMV, computed once)
//   * layer4 output never materialized: GEMM fused with global max-pool
//     via uint-bit atomicMax (post-ReLU values are >= 0).
//   * GEMM: 2x2 register tiling -> 4 v_wmma_f32_16x16x32_bf16 per k-step,
//     each A/B fragment reused twice (compute-bound on the matrix pipe).
// ---------------------------------------------------------------------------

typedef unsigned short u16;
typedef unsigned int   u32;
typedef __bf16 bf16;
typedef __attribute__((ext_vector_type(16))) bf16  v16bf;
typedef __attribute__((ext_vector_type(8)))  float v8f;
typedef __attribute__((ext_vector_type(8)))  u16   v8u16;
typedef __attribute__((ext_vector_type(16))) u16   v16u16;

__device__ __forceinline__ u16 f2bf(float f) {
  u32 u = __float_as_uint(f);
  u += 0x7FFFu + ((u >> 16) & 1u);     // round-to-nearest-even
  return (u16)(u >> 16);
}
__device__ __forceinline__ float bf2f(u16 h) {
  return __uint_as_float(((u32)h) << 16);
}

__device__ __forceinline__ v16bf load_frag(const u16* __restrict__ p) {
  // lanes 0-15: K {0..7, 16..23}; lanes 16-31: K {8..15, 24..31}
  v8u16 lo = *(const v8u16*)(p);
  v8u16 hi = *(const v8u16*)(p + 16);
  v16u16 t;
#pragma unroll
  for (int i = 0; i < 8; ++i) { t[i] = lo[i]; t[8 + i] = hi[i]; }
  return __builtin_bit_cast(v16bf, t);
}

// ---------------------------------------------------------------------------
// Tiled WMMA GEMM:  OUT[n, COUT] = act(IN[n, CIN] @ W[COUT, CIN]^T + bias)
// Block: 128 threads = 4 wave32; each wave owns a 32-row strip (2 m-tiles)
// and walks COUT in steps of 32 (2 n-tiles) -> 4 accumulators, 4 WMMA/k-step.
// A tile (128 x CIN) staged in dynamic LDS; B read from L2-resident weights.
// ---------------------------------------------------------------------------
template <int CIN, int COUT, bool STORE, bool MAXPOOL>
__global__ __launch_bounds__(128) void gemm_bf16_relu(
    const u16* __restrict__ IN, const u16* __restrict__ W,
    const float* __restrict__ bias, u16* __restrict__ OUT,
    float* __restrict__ gmax, int relu)
{
  extern __shared__ u16 smem[];                       // [128][CIN] bf16 bits
  const int tid  = threadIdx.x;
  const int lane = tid & 31;
  const int wave = tid >> 5;                          // 0..3
  const int m    = lane & 15;
  const int half = lane >> 4;                         // K-half select
  const long rowBase = (long)blockIdx.x * 128;

  // stage A tile (128 x CIN) into LDS with 16B vectors
  {
    const v8u16* __restrict__ src = (const v8u16*)(IN + rowBase * CIN);
    v8u16* dst = (v8u16*)smem;
    const int nvec = (128 * CIN) / 8;
    for (int i = tid; i < nvec; i += 128) dst[i] = src[i];
  }
  __syncthreads();

  const u16* arow0 = smem + (wave * 32 + m) * CIN + half * 8;
  const u16* arow1 = arow0 + 16 * CIN;

  for (int nt = 0; nt < COUT / 32; ++nt) {
    const u16* wrow0 = W + (long)(nt * 32 + m) * CIN + half * 8;
    const u16* wrow1 = wrow0 + 16 * CIN;
    v8f acc[2][2] = {};
#pragma unroll 4
    for (int kt = 0; kt < CIN / 32; ++kt) {
      __builtin_prefetch(wrow0 + 32 * CIN + kt * 32, 0, 1);  // next n-step's B
      v16bf a0 = load_frag(arow0 + kt * 32);
      v16bf a1 = load_frag(arow1 + kt * 32);
      v16bf b0 = load_frag(wrow0 + kt * 32);
      v16bf b1 = load_frag(wrow1 + kt * 32);
      acc[0][0] = __builtin_amdgcn_wmma_f32_16x16x32_bf16(
          false, a0, false, b0, (short)0, acc[0][0], false, false);
      acc[0][1] = __builtin_amdgcn_wmma_f32_16x16x32_bf16(
          false, a0, false, b1, (short)0, acc[0][1], false, false);
      acc[1][0] = __builtin_amdgcn_wmma_f32_16x16x32_bf16(
          false, a1, false, b0, (short)0, acc[1][0], false, false);
      acc[1][1] = __builtin_amdgcn_wmma_f32_16x16x32_bf16(
          false, a1, false, b1, (short)0, acc[1][1], false, false);
    }
    float mx[2] = {0.f, 0.f};
#pragma unroll
    for (int mt = 0; mt < 2; ++mt) {
#pragma unroll
      for (int ntl = 0; ntl < 2; ++ntl) {
        const int col = nt * 32 + ntl * 16 + m;        // output channel
        const float bb = bias[col];
#pragma unroll
        for (int r = 0; r < 8; ++r) {
          float v = acc[mt][ntl][r] + bb;
          if (relu) v = fmaxf(v, 0.f);
          if (MAXPOOL) mx[ntl] = fmaxf(mx[ntl], v);
          if (STORE) {
            long row = rowBase + wave * 32 + mt * 16 + half * 8 + r;
            ((bf16*)OUT)[row * COUT + col] = (bf16)v;  // hw cvt f32->bf16
          }
        }
      }
    }
    if (MAXPOOL) {                                     // relu>=0: bits ordered
      atomicMax((u32*)(gmax + nt * 32 + m),      __float_as_uint(mx[0]));
      atomicMax((u32*)(gmax + nt * 32 + 16 + m), __float_as_uint(mx[1]));
    }
  }
}

// --------------------------- prep / small kernels ---------------------------
__global__ void k_convert(const float* __restrict__ s, u16* __restrict__ d, int n) {
  int i = blockIdx.x * blockDim.x + threadIdx.x;
  if (i < n) d[i] = f2bf(s[i]);
}
__global__ void k_padx(const float* __restrict__ x, u16* __restrict__ xp, int npts) {
  int i = blockIdx.x * blockDim.x + threadIdx.x;
  if (i >= npts) return;
#pragma unroll
  for (int c = 0; c < 32; ++c)
    xp[(long)i * 32 + c] = (c < 3) ? f2bf(x[(long)i * 3 + c]) : (u16)0;
}
__global__ void k_padw1(const float* __restrict__ W1, u16* __restrict__ W1p) {
  int i = blockIdx.x * blockDim.x + threadIdx.x;
  if (i >= 64 * 32) return;
  int r = i >> 5, c = i & 31;
  W1p[i] = (c < 3) ? f2bf(W1[r * 3 + c]) : (u16)0;
}
__global__ void k_w5a(const float* __restrict__ W5, u16* __restrict__ W5a) {
  int i = blockIdx.x * blockDim.x + threadIdx.x;
  if (i >= 512 * 64) return;
  int r = i >> 6, c = i & 63;
  W5a[i] = f2bf(W5[(long)r * 1088 + c]);
}
__global__ void k_zero(float* __restrict__ g, int n) {
  int i = blockIdx.x * blockDim.x + threadIdx.x;
  if (i < n) g[i] = 0.f;
}
// c5[o] = b5[o] + W5[o, 64:1088] . g   (fp32, one-shot)
__global__ void k_c5(const float* __restrict__ W5, const float* __restrict__ b5,
                     const float* __restrict__ g, float* __restrict__ c5) {
  int o = blockIdx.x * blockDim.x + threadIdx.x;
  if (o >= 512) return;
  const float* wr = W5 + (long)o * 1088 + 64;
  float s = b5[o];
  for (int j = 0; j < 1024; ++j) s += wr[j] * g[j];
  c5[o] = s;
}
// layer 8: out[i] = Z7[i,:] . W8 + b8   (Cout = 1, fp32 out)
__global__ void k_l8(const u16* __restrict__ Z7, const float* __restrict__ W8,
                     const float* __restrict__ b8, float* __restrict__ out, int npts) {
  int i = blockIdx.x * blockDim.x + threadIdx.x;
  if (i >= npts) return;
  const u16* z = Z7 + (long)i * 128;
  float s = b8[0];
#pragma unroll 8
  for (int k = 0; k < 128; ++k) s += bf2f(z[k]) * W8[k];
  out[i] = s;
}

// ---------------------------------------------------------------------------
extern "C" void kernel_launch(void* const* d_in, const int* in_sizes, int n_in,
                              void* d_out, int out_size, void* d_ws, size_t ws_size,
                              hipStream_t stream) {
  (void)n_in; (void)out_size; (void)ws_size;
  const float* x  = (const float*)d_in[0];
  const float* W1 = (const float*)d_in[1];  const float* b1 = (const float*)d_in[2];
  const float* W2 = (const float*)d_in[3];  const float* b2 = (const float*)d_in[4];
  const float* W3 = (const float*)d_in[5];  const float* b3 = (const float*)d_in[6];
  const float* W4 = (const float*)d_in[7];  const float* b4 = (const float*)d_in[8];
  const float* W5 = (const float*)d_in[9];  const float* b5 = (const float*)d_in[10];
  const float* W6 = (const float*)d_in[11]; const float* b6 = (const float*)d_in[12];
  const float* W7 = (const float*)d_in[13]; const float* b7 = (const float*)d_in[14];
  const float* W8 = (const float*)d_in[15]; const float* b8 = (const float*)d_in[16];

  const int N = in_sizes[0] / 3;            // 65536 (multiple of 128)

  // workspace carve-up (256B aligned)
  char* base = (char*)d_ws;
  size_t off = 0;
  auto carve = [&](size_t bytes) -> char* {
    char* p = base + off;
    off = (off + bytes + 255) & ~(size_t)255;
    return p;
  };
  u16* Xp  = (u16*)carve((size_t)N * 32  * 2);
  u16* H1  = (u16*)carve((size_t)N * 64  * 2);
  u16* H2  = (u16*)carve((size_t)N * 64  * 2);
  u16* Y3  = (u16*)carve((size_t)N * 128 * 2);
  u16* Z5  = (u16*)carve((size_t)N * 512 * 2);
  u16* Z6  = (u16*)carve((size_t)N * 256 * 2);
  u16* Z7  = (u16*)carve((size_t)N * 128 * 2);
  u16* W1p = (u16*)carve(64   * 32  * 2);
  u16* W2b = (u16*)carve(64   * 64  * 2);
  u16* W3b = (u16*)carve(128  * 64  * 2);
  u16* W4b = (u16*)carve(1024 * 128 * 2);
  u16* W5a = (u16*)carve(512  * 64  * 2);
  u16* W6b = (u16*)carve(256  * 512 * 2);
  u16* W7b = (u16*)carve(128  * 256 * 2);
  float* g  = (float*)carve(1024 * 4);
  float* c5 = (float*)carve(512 * 4);

  auto cvt = [&](const float* s, u16* d, int n) {
    k_convert<<<(n + 255) / 256, 256, 0, stream>>>(s, d, n);
  };

  // ---- prep: bf16 weight conversion + input padding + g = 0 ----
  k_padx <<<(N + 255) / 256, 256, 0, stream>>>(x, Xp, N);
  k_padw1<<<8, 256, 0, stream>>>(W1, W1p);
  cvt(W2, W2b, 64 * 64);
  cvt(W3, W3b, 128 * 64);
  cvt(W4, W4b, 1024 * 128);
  k_w5a<<<(512 * 64 + 255) / 256, 256, 0, stream>>>(W5, W5a);
  cvt(W6, W6b, 256 * 512);
  cvt(W7, W7b, 128 * 256);
  k_zero<<<4, 256, 0, stream>>>(g, 1024);

  const int blocks = N / 128;               // 128 rows per block, 4 waves

  // ---- layers 1..4 (layer4 fused with global max-pool, no store) ----
  gemm_bf16_relu< 32,   64, true,  false><<<blocks, 128, 128 *  32 * 2, stream>>>(Xp, W1p, b1, H1, nullptr, 1);
  gemm_bf16_relu< 64,   64, true,  false><<<blocks, 128, 128 *  64 * 2, stream>>>(H1, W2b, b2, H2, nullptr, 1);
  gemm_bf16_relu< 64,  128, true,  false><<<blocks, 128, 128 *  64 * 2, stream>>>(H2, W3b, b3, Y3, nullptr, 1);
  gemm_bf16_relu<128, 1024, false, true ><<<blocks, 128, 128 * 128 * 2, stream>>>(Y3, W4b, b4, nullptr, g, 1);

  // ---- fold global feature through W5's concat half ----
  k_c5<<<2, 256, 0, stream>>>(W5, b5, g, c5);

  // ---- layers 5..7 ----
  gemm_bf16_relu< 64,  512, true, false><<<blocks, 128, 128 *  64 * 2, stream>>>(H2, W5a, c5, Z5, nullptr, 1);
  gemm_bf16_relu<512,  256, true, false><<<blocks, 128, 128 * 512 * 2, stream>>>(Z5, W6b, b6, Z6, nullptr, 1);
  gemm_bf16_relu<256,  128, true, false><<<blocks, 128, 128 * 256 * 2, stream>>>(Z6, W7b, b7, Z7, nullptr, 1);

  // ---- layer 8 (no relu, fp32 output) ----
  k_l8<<<(N + 255) / 256, 256, 0, stream>>>(Z7, W8, b8, (float*)d_out, N);
}
